// PathMoE_33234456936754
// MI455X (gfx1250) — compile-verified, matrix operand order checked by reading, not run on hardware
//
#include <hip/hip_runtime.h>
#include <hip/hip_bf16.h>
#include <math.h>

typedef __attribute__((ext_vector_type(16))) __bf16 v16bf;
typedef __attribute__((ext_vector_type(8)))  float  v8f;

#define N_INST 32768
#define L_DIM  1024
#define D1     512
#define D2     256
#define NBAG   4
#define KSEL   8
#define CHUNKS 64
#define ROWS_TOTAL (NBAG * N_INST)   // 131072

// ---------------- workspace layout (bytes) ----------------
#define WS_H      0ull                                  // bf16 [131072][512]
#define WS_WFC    (WS_H     + (size_t)ROWS_TOTAL*D1*2)  // bf16 [512][1024]   W_fc^T
#define WS_WA     (WS_WFC   + (size_t)D1*L_DIM*2)       // bf16 [256][512]    Wa^T
#define WS_WB     (WS_WA    + (size_t)D2*D1*2)          // bf16 [256][512]    Wb^T
#define WS_A      (WS_WB    + (size_t)D2*D1*2)          // f32  [4][32768]    softmax(A_raw)
#define WS_MPART  (WS_A     + (size_t)NBAG*N_INST*4)    // f32  [64][4][512]
#define WS_M      (WS_MPART + (size_t)CHUNKS*NBAG*D1*4) // f32  [4][512]
#define WS_IDX    (WS_M     + (size_t)NBAG*D1*4)        // int  [4][16]
#define WS_LOSSP  (WS_IDX   + (size_t)NBAG*16*4)        // f32  [64]

// ---------------- output layout (float elements) ----------------
#define OUT_LOGITS 0
#define OUT_YPROB  2
#define OUT_YHAT   4
#define OUT_ARAW   5
#define OUT_LOSS   (OUT_ARAW + ROWS_TOTAL)  // 131077
#define OUT_PREDS  (OUT_LOSS + 1)           // 131078

__device__ __forceinline__ v8f wmma_bf16(v16bf a, v16bf b, v8f c) {
  return __builtin_amdgcn_wmma_f32_16x16x32_bf16(false, a, false, b, (short)0, c,
                                                 false, false);
}

// ============ prep: transpose + fp32->bf16 convert: out[n*K+k] = in[k*N+n] ============
__global__ void k_transpose_cvt(const float* __restrict__ in, __bf16* __restrict__ out,
                                int K, int Ncol) {
  int i = blockIdx.x * blockDim.x + threadIdx.x;
  int total = K * Ncol;
  if (i >= total) return;
  int n = i / K;
  int k = i - n * K;
  out[i] = (__bf16)in[(size_t)k * Ncol + n];
}

// ============ GEMM 1: h = relu(X @ W_fc + b_fc), stored bf16 ============
// grid (2048, 4), block 256 (8 waves). Each block: 16 rows x 512 cols, K=1024.
__global__ __launch_bounds__(256)
void k_gemm1_relu(const float* __restrict__ x0, const float* __restrict__ x1,
                  const float* __restrict__ x2, const float* __restrict__ x3,
                  const __bf16* __restrict__ wfc_t, const float* __restrict__ bfc,
                  __bf16* __restrict__ hout) {
  const int bag = blockIdx.y;
  const float* X = (bag == 0) ? x0 : (bag == 1) ? x1 : (bag == 2) ? x2 : x3;
  const int rowbase = blockIdx.x * 16;
  const int tid = threadIdx.x;
  const int w = tid >> 5, l = tid & 31;
  const int lrow = l & 15, lhalf = l >> 4;
  const int nb0 = w * 64;

  v8f acc0 = {}, acc1 = {}, acc2 = {}, acc3 = {};
  const float* arow = X + (size_t)(rowbase + lrow) * L_DIM + lhalf * 16;
  const __bf16* bbase = wfc_t + (size_t)(nb0 + lrow) * L_DIM + lhalf * 16;

  for (int kb = 0; kb < L_DIM; kb += 32) {
    __builtin_prefetch(arow + kb + 32, 0, 0);   // global_prefetch_b8
    const float4* ap = (const float4*)(arow + kb);
    float4 f0 = ap[0], f1 = ap[1], f2 = ap[2], f3 = ap[3];
    v16bf af;
    af[0]=(__bf16)f0.x; af[1]=(__bf16)f0.y; af[2]=(__bf16)f0.z; af[3]=(__bf16)f0.w;
    af[4]=(__bf16)f1.x; af[5]=(__bf16)f1.y; af[6]=(__bf16)f1.z; af[7]=(__bf16)f1.w;
    af[8]=(__bf16)f2.x; af[9]=(__bf16)f2.y; af[10]=(__bf16)f2.z; af[11]=(__bf16)f2.w;
    af[12]=(__bf16)f3.x; af[13]=(__bf16)f3.y; af[14]=(__bf16)f3.z; af[15]=(__bf16)f3.w;

    const __bf16* bp = bbase + kb;
    v16bf b0 = *(const v16bf*)(bp);
    v16bf b1 = *(const v16bf*)(bp + 16 * L_DIM);
    v16bf b2 = *(const v16bf*)(bp + 32 * L_DIM);
    v16bf b3 = *(const v16bf*)(bp + 48 * L_DIM);
    acc0 = wmma_bf16(af, b0, acc0);
    acc1 = wmma_bf16(af, b1, acc1);
    acc2 = wmma_bf16(af, b2, acc2);
    acc3 = wmma_bf16(af, b3, acc3);
  }

  // epilogue: bias + relu + bf16 store. lane holds (m = r + 8*lhalf, n = nt + lrow)
  const size_t rowg0 = (size_t)(bag * N_INST + rowbase);
#pragma unroll
  for (int t = 0; t < 4; ++t) {
    const v8f acc = (t == 0) ? acc0 : (t == 1) ? acc1 : (t == 2) ? acc2 : acc3;
    const int n = nb0 + t * 16 + lrow;
    const float bv = bfc[n];
#pragma unroll
    for (int r = 0; r < 8; ++r) {
      const int m = r + 8 * lhalf;
      float v = fmaxf(acc[r] + bv, 0.0f);
      hout[(rowg0 + m) * D1 + n] = (__bf16)v;
    }
  }
}

// ============ GEMM 2+3 fused: A_raw = (tanh(h@Wa+ba)*sigmoid(h@Wb+bb)) @ Wc + bc ============
// grid 8192, block 256 (8 waves). Each block: 16 rows; each wave 32 cols of both Wa & Wb.
__global__ __launch_bounds__(256)
void k_gemm2_attn(const __bf16* __restrict__ h,
                  const __bf16* __restrict__ wa_t, const __bf16* __restrict__ wb_t,
                  const float* __restrict__ ba, const float* __restrict__ bb,
                  const float* __restrict__ wc, const float* __restrict__ bc,
                  float* __restrict__ araw) {
  __shared__ float s_part[8][16];
  const int rowbase = blockIdx.x * 16;
  const int tid = threadIdx.x;
  const int w = tid >> 5, l = tid & 31;
  const int lrow = l & 15, lhalf = l >> 4;
  const int nb0 = w * 32;

  v8f ca0 = {}, ca1 = {}, cb0 = {}, cb1 = {};
  const __bf16* arow = h + (size_t)(rowbase + lrow) * D1 + lhalf * 16;
  const __bf16* bpa0 = wa_t + (size_t)(nb0 + lrow) * D1 + lhalf * 16;
  const __bf16* bpb0 = wb_t + (size_t)(nb0 + lrow) * D1 + lhalf * 16;

  for (int kb = 0; kb < D1; kb += 32) {
    v16bf af = *(const v16bf*)(arow + kb);
    v16bf a0 = *(const v16bf*)(bpa0 + kb);
    v16bf a1 = *(const v16bf*)(bpa0 + kb + 16 * D1);
    v16bf g0 = *(const v16bf*)(bpb0 + kb);
    v16bf g1 = *(const v16bf*)(bpb0 + kb + 16 * D1);
    ca0 = wmma_bf16(af, a0, ca0);
    ca1 = wmma_bf16(af, a1, ca1);
    cb0 = wmma_bf16(af, g0, cb0);
    cb1 = wmma_bf16(af, g1, cb1);
  }

  float rs[8];
#pragma unroll
  for (int r = 0; r < 8; ++r) rs[r] = 0.0f;
#pragma unroll
  for (int t = 0; t < 2; ++t) {
    const int n = nb0 + t * 16 + lrow;
    const float bav = ba[n], bbv = bb[n], wcv = wc[n];
    const v8f A_ = t ? ca1 : ca0;
    const v8f G_ = t ? cb1 : cb0;
#pragma unroll
    for (int r = 0; r < 8; ++r) {
      float av = tanhf(A_[r] + bav);
      float gv = 1.0f / (1.0f + expf(-(G_[r] + bbv)));
      rs[r] += av * gv * wcv;
    }
  }
  // reduce across the 16 lanes of each half-wave (different n, same m set)
#pragma unroll
  for (int off = 1; off < 16; off <<= 1) {
#pragma unroll
    for (int r = 0; r < 8; ++r) rs[r] += __shfl_xor(rs[r], off, 16);
  }
  if (lrow == 0) {
#pragma unroll
    for (int r = 0; r < 8; ++r) s_part[w][lhalf * 8 + r] = rs[r];
  }
  __syncthreads();
  if (tid < 16) {
    float s = bc[0];
#pragma unroll
    for (int ww = 0; ww < 8; ++ww) s += s_part[ww][tid];   // fixed order -> deterministic
    araw[rowbase + tid] = s;
  }
}

// ============ softmax over each bag's 32768 scores ============
__global__ __launch_bounds__(256)
void k_softmax(const float* __restrict__ araw, float* __restrict__ Aout) {
  const int b = blockIdx.x, tid = threadIdx.x;
  const float* x = araw + (size_t)b * N_INST;
  float* y = Aout + (size_t)b * N_INST;
  __shared__ float red[256];
  float mx = -INFINITY;
  for (int i = tid; i < N_INST; i += 256) mx = fmaxf(mx, x[i]);
  red[tid] = mx; __syncthreads();
  for (int s = 128; s > 0; s >>= 1) { if (tid < s) red[tid] = fmaxf(red[tid], red[tid + s]); __syncthreads(); }
  const float m = red[0]; __syncthreads();
  float sum = 0.0f;
  for (int i = tid; i < N_INST; i += 256) sum += expf(x[i] - m);
  red[tid] = sum; __syncthreads();
  for (int s = 128; s > 0; s >>= 1) { if (tid < s) red[tid] += red[tid + s]; __syncthreads(); }
  const float inv = 1.0f / red[0];
  for (int i = tid; i < N_INST; i += 256) y[i] = expf(x[i] - m) * inv;
}

// ============ top-8 / bottom-8 per bag (lowest-index tie break, like lax.top_k) ============
__global__ __launch_bounds__(256)
void k_topk(const float* __restrict__ A, int* __restrict__ idxout) {
  const int b = blockIdx.x, tid = threadIdx.x;
  const float* s = A + (size_t)b * N_INST;
  __shared__ unsigned char taken[N_INST];   // 32 KB LDS
  __shared__ float rv[256];
  __shared__ int   ri[256];
  for (int i = tid; i < N_INST; i += 256) taken[i] = 0;
  __syncthreads();
  for (int iter = 0; iter < KSEL; ++iter) {           // top-8
    float bv = -INFINITY; int bi = N_INST;
    for (int i = tid; i < N_INST; i += 256) {
      if (taken[i]) continue;
      float v = s[i];
      if (v > bv || (v == bv && i < bi)) { bv = v; bi = i; }
    }
    rv[tid] = bv; ri[tid] = bi; __syncthreads();
    for (int st = 128; st > 0; st >>= 1) {
      if (tid < st) {
        float ov = rv[tid + st]; int oi = ri[tid + st];
        if (ov > rv[tid] || (ov == rv[tid] && oi < ri[tid])) { rv[tid] = ov; ri[tid] = oi; }
      }
      __syncthreads();
    }
    if (tid == 0) { idxout[b * 16 + iter] = ri[0]; taken[ri[0]] = 1; }
    __syncthreads();
  }
  for (int i = tid; i < N_INST; i += 256) taken[i] = 0;
  __syncthreads();
  for (int iter = 0; iter < KSEL; ++iter) {           // bottom-8 (= top_k of -s)
    float bv = INFINITY; int bi = N_INST;
    for (int i = tid; i < N_INST; i += 256) {
      if (taken[i]) continue;
      float v = s[i];
      if (v < bv || (v == bv && i < bi)) { bv = v; bi = i; }
    }
    rv[tid] = bv; ri[tid] = bi; __syncthreads();
    for (int st = 128; st > 0; st >>= 1) {
      if (tid < st) {
        float ov = rv[tid + st]; int oi = ri[tid + st];
        if (ov < rv[tid] || (ov == rv[tid] && oi < ri[tid])) { rv[tid] = ov; ri[tid] = oi; }
      }
      __syncthreads();
    }
    if (tid == 0) { idxout[b * 16 + 8 + iter] = ri[0]; taken[ri[0]] = 1; }
    __syncthreads();
  }
}

// ============ instance logits, CE parts, predictions ============
__global__ __launch_bounds__(128)
void k_inst(const __bf16* __restrict__ h, const int* __restrict__ idx,
            const float* __restrict__ instW, const float* __restrict__ instB,
            float* __restrict__ lossparts, float* __restrict__ preds_out) {
  const int blk = blockIdx.x;             // b*16 + j
  const int b = blk >> 4, j = blk & 15;
  const int tid = threadIdx.x;
  const size_t row = (size_t)b * N_INST + idx[b * 16 + j];
  const __bf16* hr = h + row * D1;
  __shared__ float r0[128], r1[128];
  float s0 = 0.0f, s1 = 0.0f;
  for (int d = tid; d < D1; d += 128) {
    float hv = (float)hr[d];
    s0 += hv * instW[d * 2 + 0];
    s1 += hv * instW[d * 2 + 1];
  }
  r0[tid] = s0; r1[tid] = s1; __syncthreads();
  for (int st = 64; st > 0; st >>= 1) {
    if (tid < st) { r0[tid] += r0[tid + st]; r1[tid] += r1[tid + st]; }
    __syncthreads();
  }
  if (tid == 0) {
    float l0 = r0[0] + instB[0], l1 = r1[0] + instB[1];
    float m = fmaxf(l0, l1);
    float lse = m + logf(expf(l0 - m) + expf(l1 - m));
    float lptgt = ((j < KSEL) ? l1 : l0) - lse;
    lossparts[blk] = -lptgt * (1.0f / 16.0f);
    preds_out[blk] = (l1 > l0) ? 1.0f : 0.0f;   // argmax, first-index tie break
  }
}

// ============ attention pooling M = A @ h, two-stage deterministic reduce ============
__global__ __launch_bounds__(512)
void k_mpool_partial(const __bf16* __restrict__ h, const float* __restrict__ A,
                     float* __restrict__ mpart) {
  const int c = blockIdx.x, b = blockIdx.y, d = threadIdx.x;
  const int rpb = N_INST / CHUNKS;     // 512 rows per chunk
  const int r0 = c * rpb;
  const __bf16* hb = h + (size_t)b * N_INST * D1;
  const float* Ab = A + (size_t)b * N_INST;
  float acc = 0.0f;
  for (int r = r0; r < r0 + rpb; ++r) acc += Ab[r] * (float)hb[(size_t)r * D1 + d];
  mpart[((size_t)c * NBAG + b) * D1 + d] = acc;
}

__global__ __launch_bounds__(512)
void k_mpool_reduce(const float* __restrict__ mpart, float* __restrict__ M) {
  const int b = blockIdx.x, d = threadIdx.x;
  float s = 0.0f;
  for (int c = 0; c < CHUNKS; ++c) s += mpart[((size_t)c * NBAG + b) * D1 + d];
  M[b * D1 + d] = s;
}

// ============ final heads: bag attention, classifier, outputs ============
__global__ __launch_bounds__(512)
void k_final(const float* __restrict__ M, const float* __restrict__ attnW,
             const float* __restrict__ attnB, const float* __restrict__ clsW,
             const float* __restrict__ clsB, const float* __restrict__ lossparts,
             float* __restrict__ out) {
  const int tid = threadIdx.x;
  __shared__ float sM[NBAG * D1];       // 2048
  __shared__ float w4[NBAG];
  __shared__ float Mf[D1];
  __shared__ float lg[2];
  for (int i = tid; i < NBAG * D1; i += 512) sM[i] = M[i];
  __syncthreads();
  if (tid < NBAG) {
    float s = attnB[tid];
    for (int i = 0; i < NBAG * D1; ++i) s += sM[i] * attnW[i * NBAG + tid];
    w4[tid] = s;
  }
  __syncthreads();
  if (tid == 0) {
    float m = fmaxf(fmaxf(w4[0], w4[1]), fmaxf(w4[2], w4[3]));
    float e0 = expf(w4[0] - m), e1 = expf(w4[1] - m),
          e2 = expf(w4[2] - m), e3 = expf(w4[3] - m);
    float inv = 1.0f / (e0 + e1 + e2 + e3);
    w4[0] = e0 * inv; w4[1] = e1 * inv; w4[2] = e2 * inv; w4[3] = e3 * inv;
  }
  __syncthreads();
  if (tid < D1) {
    float s = 0.0f;
    for (int b = 0; b < NBAG; ++b) s += w4[b] * sM[b * D1 + tid];
    Mf[tid] = s;
  }
  __syncthreads();
  if (tid < 2) {
    float s = clsB[tid];
    for (int d = 0; d < D1; ++d) s += Mf[d] * clsW[d * 2 + tid];
    lg[tid] = s;
  }
  __syncthreads();
  if (tid == 0) {
    float l0 = lg[0], l1 = lg[1];
    out[OUT_LOGITS + 0] = l0;
    out[OUT_LOGITS + 1] = l1;
    float m = fmaxf(l0, l1);
    float e0 = expf(l0 - m), e1 = expf(l1 - m);
    float inv = 1.0f / (e0 + e1);
    out[OUT_YPROB + 0] = e0 * inv;
    out[OUT_YPROB + 1] = e1 * inv;
    out[OUT_YHAT] = (l1 > l0) ? 1.0f : 0.0f;
    float tl = 0.0f;
    for (int i = 0; i < NBAG * 16; ++i) tl += lossparts[i];  // fixed order
    out[OUT_LOSS] = tl;
  }
}

extern "C" void kernel_launch(void* const* d_in, const int* in_sizes, int n_in,
                              void* d_out, int out_size, void* d_ws, size_t ws_size,
                              hipStream_t stream) {
  (void)in_sizes; (void)n_in; (void)out_size; (void)ws_size;
  const float* h1   = (const float*)d_in[0];
  const float* h2   = (const float*)d_in[1];
  const float* h3   = (const float*)d_in[2];
  const float* h4   = (const float*)d_in[3];
  const float* Wfc  = (const float*)d_in[4];
  const float* bfc  = (const float*)d_in[5];
  const float* Wa   = (const float*)d_in[6];
  const float* ba   = (const float*)d_in[7];
  const float* Wb   = (const float*)d_in[8];
  const float* bb   = (const float*)d_in[9];
  const float* Wc   = (const float*)d_in[10];
  const float* bc   = (const float*)d_in[11];
  const float* iW   = (const float*)d_in[12];
  const float* iB   = (const float*)d_in[13];
  const float* aW   = (const float*)d_in[14];
  const float* aB   = (const float*)d_in[15];
  const float* cW   = (const float*)d_in[16];
  const float* cB   = (const float*)d_in[17];
  float* out = (float*)d_out;

  char* ws = (char*)d_ws;
  __bf16* h_bf   = (__bf16*)(ws + WS_H);
  __bf16* wfc_t  = (__bf16*)(ws + WS_WFC);
  __bf16* wa_t   = (__bf16*)(ws + WS_WA);
  __bf16* wb_t   = (__bf16*)(ws + WS_WB);
  float*  A_sm   = (float*)(ws + WS_A);
  float*  mpart  = (float*)(ws + WS_MPART);
  float*  Mws    = (float*)(ws + WS_M);
  int*    idxws  = (int*)(ws + WS_IDX);
  float*  losspw = (float*)(ws + WS_LOSSP);

  // 0) weight transpose + bf16 convert
  k_transpose_cvt<<<(L_DIM * D1 + 255) / 256, 256, 0, stream>>>(Wfc, wfc_t, L_DIM, D1);
  k_transpose_cvt<<<(D1 * D2 + 255) / 256, 256, 0, stream>>>(Wa, wa_t, D1, D2);
  k_transpose_cvt<<<(D1 * D2 + 255) / 256, 256, 0, stream>>>(Wb, wb_t, D1, D2);

  // 1) h = relu(X @ W_fc + b) via bf16 WMMA
  k_gemm1_relu<<<dim3(N_INST / 16, NBAG), 256, 0, stream>>>(h1, h2, h3, h4, wfc_t, bfc, h_bf);

  // 2+3+4) fused gated attention scores -> A_raw (written directly to output slot)
  k_gemm2_attn<<<ROWS_TOTAL / 16, 256, 0, stream>>>(h_bf, wa_t, wb_t, ba, bb, Wc, bc,
                                                    out + OUT_ARAW);

  // 5) softmax per bag
  k_softmax<<<NBAG, 256, 0, stream>>>(out + OUT_ARAW, A_sm);

  // 6) top-8 / bottom-8 indices per bag
  k_topk<<<NBAG, 256, 0, stream>>>(A_sm, idxws);

  // 7) instance logits, loss parts, predictions
  k_inst<<<NBAG * 16, 128, 0, stream>>>(h_bf, idxws, iW, iB, losspw, out + OUT_PREDS);

  // 8) M = A @ h (two-stage deterministic reduction)
  k_mpool_partial<<<dim3(CHUNKS, NBAG), 512, 0, stream>>>(h_bf, A_sm, mpart);
  k_mpool_reduce<<<NBAG, 512, 0, stream>>>(mpart, Mws);

  // 9) final heads + scalar outputs
  k_final<<<1, 512, 0, stream>>>(Mws, aW, aB, cW, cB, losspw, out);
}
